// OSSBlock_80444737454938
// MI455X (gfx1250) — compile-verified, matrix operand order checked by reading, not run on hardware
//
#include <hip/hip_runtime.h>
#include <hip/hip_bf16.h>
#include <math.h>

typedef __attribute__((ext_vector_type(16))) _Float16 v16h;
typedef __attribute__((ext_vector_type(8)))  _Float16 v8h;
typedef __attribute__((ext_vector_type(8)))  float    v8f;

#define BB   2
#define CC   32
#define HH   64
#define WWD  64
#define HW   4096
#define MM   (BB*HW)      /* 8192 token rows */
#define EE   256
#define HID  64
/* mamba1 */
#define DI1   256
#define DTR1  8
#define NST   16
#define NPROJ1 40         /* dtr + 2*16 */
#define NPROJ1_PAD 48     /* padded to 16 for unguarded WMMA B loads */
/* mamba2 */
#define DI2   4

__device__ __forceinline__ float siluf(float x) { return x / (1.f + __expf(-x)); }
__device__ __forceinline__ float softplusf(float x) { return (x > 20.f) ? x : log1pf(__expf(x)); }

/* ------------- f32 -> f16 weight pack with zero tail-padding ------------- */
__global__ void pack_f16_kernel(const float* __restrict__ src, _Float16* __restrict__ dst,
                                int n_src, int n_total) {
    int i = blockIdx.x * blockDim.x + threadIdx.x;
    if (i < n_total) dst[i] = (i < n_src) ? (_Float16)src[i] : (_Float16)0.f;
}

/* ---------------- generic WMMA GEMM: C[m,n] = sum_k A[m,k]*W[n,k] + bias[n] ----------------
 * A: (M,K) f16 row-major, W: (>=gridX*16, K) f16 row-major, zero-padded past N.
 * One wave per 16x16 output tile. K multiple of 32; M multiple of 16; store guarded on N.
 * Fragment layout per ISA 7.12.2: lanes 0-15 hold K{0..7,16..23}, lanes 16-31 K{8..15,24..31};
 * D: VGPR v -> rows v (lanes 0-15) / v+8 (lanes 16-31). */
__global__ void __launch_bounds__(32)
wmma_gemm_kernel(const _Float16* __restrict__ A, const _Float16* __restrict__ Bw,
                 const float* __restrict__ bias, float* __restrict__ C,
                 int M, int N, int K)
{
    int lane  = threadIdx.x & 31;
    int mtile = blockIdx.y, ntile = blockIdx.x;
    int row   = lane & 15;
    int hi    = lane >> 4;
    int koff  = hi << 3;
    const _Float16* arow = A  + (size_t)(mtile * 16 + row) * K;
    const _Float16* brow = Bw + (size_t)(ntile * 16 + row) * K;   /* always valid (padded) */

    v8f acc;
#pragma unroll
    for (int i = 0; i < 8; ++i) acc[i] = 0.f;

    union V16 { v16h v; v8h h[2]; };
    for (int k = 0; k < K; k += 32) {
        __builtin_prefetch(arow + k + 64, 0, 1);   /* global_prefetch_b8 */
        V16 a, b;
        a.h[0] = *(const v8h*)(arow + k + koff);
        a.h[1] = *(const v8h*)(arow + k + 16 + koff);
        b.h[0] = *(const v8h*)(brow + k + koff);
        b.h[1] = *(const v8h*)(brow + k + 16 + koff);
        acc = __builtin_amdgcn_wmma_f32_16x16x32_f16(false, a.v, false, b.v,
                                                     (short)0, acc, false, false);
    }

    int cn = ntile * 16 + row;
    if (cn < N) {
        float bv = bias ? bias[cn] : 0.f;
        int mbase = mtile * 16 + (hi << 3);
#pragma unroll
        for (int v = 0; v < 8; ++v)
            C[(size_t)(mbase + v) * N + cn] = acc[v] + bv;
    }
}

/* ---------------- emb MLP: e = silu(emb) @ aw.T + ab (both adaLN heads) ---------------- */
__global__ void emb_mlp_kernel(const float* __restrict__ emb,
                               const float* __restrict__ a1_w, const float* __restrict__ a1_b,
                               const float* __restrict__ a2_w, const float* __restrict__ a2_b,
                               float* __restrict__ e1, float* __restrict__ e2)
{
    __shared__ float se[EE];
    int b = blockIdx.x, t = threadIdx.x;
    for (int i = t; i < EE; i += blockDim.x) se[i] = siluf(emb[b * EE + i]);
    __syncthreads();
    if (t < 192) {
        int which = (t >= 96);
        int j = which ? (t - 96) : t;
        const float* w  = which ? a2_w : a1_w;
        float acc = which ? a2_b[j] : a1_b[j];
        for (int k = 0; k < EE; ++k) acc += w[j * EE + k] * se[k];
        (which ? e2 : e1)[b * 96 + j] = acc;
    }
}

/* ---------------- transpose x (B,C,HW) -> (B,HW,C) ---------------- */
__global__ void transpose_kernel(const float* __restrict__ x, float* __restrict__ xt) {
    int idx = blockIdx.x * blockDim.x + threadIdx.x;
    if (idx >= BB * HW * CC) return;
    int c = idx & 31, p = (idx >> 5) & (HW - 1), b = idx >> 17;
    xt[idx] = x[((size_t)(b * CC) + c) * HW + p];
}

/* ---------------- adaLN over channels (token-major input), f16 out ---------------- */
__global__ void adaln_kernel(const float* __restrict__ xt, const float* __restrict__ e,
                             _Float16* __restrict__ out16)
{
    int p = blockIdx.x * blockDim.x + threadIdx.x;
    if (p >= MM) return;
    int b = p >> 12;
    const float* xr = xt + (size_t)p * CC;
    float mu = 0.f;
#pragma unroll
    for (int c = 0; c < CC; ++c) mu += xr[c];
    mu *= (1.f / CC);
    float var = 0.f;
#pragma unroll
    for (int c = 0; c < CC; ++c) { float d = xr[c] - mu; var += d * d; }
    var *= (1.f / CC);
    float rinv = rsqrtf(var + 1e-5f);
#pragma unroll
    for (int c = 0; c < CC; ++c) {
        float xn = (xr[c] - mu) * rinv;
        out16[(size_t)p * CC + c] =
            (_Float16)(xn * (1.f + e[b * 96 + 32 + c]) + e[b * 96 + c]);
    }
}

/* -------- depthwise 3x3 + SiLU on fo1; SiLU(fo2); scatter 4 scan directions -------- */
__global__ void dw_dir_kernel(const float* __restrict__ fo, const float* __restrict__ dw_w,
                              const float* __restrict__ dw_b,
                              _Float16* __restrict__ fop16, float* __restrict__ fo2s)
{
    int idx = blockIdx.x * blockDim.x + threadIdx.x;
    if (idx >= BB * HW * CC) return;
    int c = idx & 31, p = (idx >> 5) & (HW - 1), b = idx >> 17;
    int h = p >> 6, w = p & 63;
    float acc = dw_b[c];
#pragma unroll
    for (int dy = -1; dy <= 1; ++dy)
#pragma unroll
        for (int dx = -1; dx <= 1; ++dx) {
            int hh = h + dy, ww = w + dx;
            if (hh >= 0 && hh < HH && ww >= 0 && ww < WWD)
                acc += dw_w[c * 9 + (dy + 1) * 3 + (dx + 1)] *
                       fo[((size_t)(b * HW) + (hh << 6) + ww) * 64 + c];
        }
    float v = siluf(acc);
    size_t base = (size_t)b * HW * 128;
    int lw = w * HH + h;
    fop16[base + (size_t)p * 128 + c]                    = (_Float16)v;  /* hf */
    fop16[base + (size_t)(HW - 1 - p) * 128 + 32 + c]    = (_Float16)v;  /* hb */
    fop16[base + (size_t)lw * 128 + 64 + c]              = (_Float16)v;  /* wf */
    fop16[base + (size_t)(HW - 1 - lw) * 128 + 96 + c]   = (_Float16)v;  /* wb */
    fo2s[idx] = siluf(fo[((size_t)(b * HW) + p) * 64 + 32 + c]);
}

/* ---------------- mamba1 causal conv (D_CONV=4) + SiLU ---------------- */
__global__ void cconv1_kernel(const float* __restrict__ xz, const float* __restrict__ cw,
                              const float* __restrict__ cb,
                              float* __restrict__ xc, _Float16* __restrict__ xc16)
{
    int idx = blockIdx.x * blockDim.x + threadIdx.x;
    if (idx >= BB * HW * DI1) return;
    int d = idx & 255, l = (idx >> 8) & (HW - 1), b = idx >> 20;
    float acc = cb[d];
#pragma unroll
    for (int k = 0; k < 4; ++k) {
        int li = l - 3 + k;
        if (li >= 0) acc += cw[d * 4 + k] * xz[((size_t)(b * HW) + li) * 512 + d];
    }
    float v = siluf(acc);
    xc[idx] = v;
    xc16[idx] = (_Float16)v;
}

/* ---------------- dt = softplus(dt_raw @ dt_w.T + dt_b), K=8 ---------------- */
__global__ void dt_kernel(const float* __restrict__ proj, const float* __restrict__ dtw,
                          const float* __restrict__ dtb, float* __restrict__ dt)
{
    int idx = blockIdx.x * blockDim.x + threadIdx.x;
    if (idx >= BB * HW * DI1) return;
    int d = idx & 255, bl = idx >> 8;
    const float* pr = proj + (size_t)bl * NPROJ1;
    float acc = dtb[d];
#pragma unroll
    for (int r = 0; r < DTR1; ++r) acc += dtw[d * DTR1 + r] * pr[r];
    dt[idx] = softplusf(acc);
}

/* ------- mamba1 selective scan: block per batch, thread per d_inner channel -------
 * 16-state recurrence in VGPRs; B/C projections staged through LDS in chunks of 64. */
#define LCHK 64
__global__ void __launch_bounds__(256)
mamba1_scan_kernel(const float* __restrict__ proj, const float* __restrict__ dt,
                   const float* __restrict__ xc, const float* __restrict__ xz,
                   const float* __restrict__ A_log, const float* __restrict__ Dp,
                   _Float16* __restrict__ y16)
{
    int b = blockIdx.x;
    int d = threadIdx.x;
    __shared__ float sB[LCHK][NST];
    __shared__ float sC[LCHK][NST];
    float A[NST], h[NST];
#pragma unroll
    for (int s = 0; s < NST; ++s) { A[s] = -__expf(A_log[d * NST + s]); h[s] = 0.f; }
    float Dv = Dp[d];
    const float* projb = proj + (size_t)b * HW * NPROJ1;

    for (int l0 = 0; l0 < HW; l0 += LCHK) {
        __syncthreads();
        for (int i = d; i < LCHK * 32; i += 256) {
            int r = i >> 5, col = i & 31;
            float v = projb[(size_t)(l0 + r) * NPROJ1 + 8 + col];
            if (col < 16) sB[r][col] = v; else sC[r][col - 16] = v;
        }
        __syncthreads();
        for (int j = 0; j < LCHK; ++j) {
            int l = l0 + j;
            size_t idx = ((size_t)(b * HW) + l) * DI1 + d;
            float dtv = dt[idx];
            float xcv = xc[idx];
            float zv  = xz[((size_t)(b * HW) + l) * 512 + 256 + d];
            float y = 0.f;
#pragma unroll
            for (int s = 0; s < NST; ++s) {
                float dA = __expf(dtv * A[s]);
                h[s] = dA * h[s] + dtv * sB[j][s] * xcv;
                y += h[s] * sC[j][s];
            }
            y16[idx] = (_Float16)((y + Dv * xcv) * siluf(zv));
        }
    }
}

/* ---------------- sum 4 directions at same seq index, multiply by SiLU(fo2) ---------------- */
__global__ void dirsum_kernel(const float* __restrict__ mam1, const float* __restrict__ fo2s,
                              float* __restrict__ res)
{
    int idx = blockIdx.x * blockDim.x + threadIdx.x;
    if (idx >= BB * HW * CC) return;
    int c = idx & 31, p = (idx >> 5) & (HW - 1), b = idx >> 17;
    const float* mo = mam1 + ((size_t)(b * HW) + p) * 128;
    res[idx] = (mo[c] + mo[32 + c] + mo[64 + c] + mo[96 + c]) * fo2s[idx];
}

/* ---------------- spatial mean -> foc (B,C) ---------------- */
__global__ void chmean_kernel(const float* __restrict__ res, float* __restrict__ foc) {
    __shared__ float sm[128];
    int b = blockIdx.x >> 5, c = blockIdx.x & 31;
    float acc = 0.f;
    for (int p = threadIdx.x; p < HW; p += 128)
        acc += res[((size_t)(b * HW) + p) * CC + c];
    sm[threadIdx.x] = acc;
    __syncthreads();
    for (int s = 64; s > 0; s >>= 1) {
        if ((int)threadIdx.x < s) sm[threadIdx.x] += sm[threadIdx.x + s];
        __syncthreads();
    }
    if (threadIdx.x == 0) foc[b * CC + c] = sm[0] * (1.f / HW);
}

/* ---------------- tiny channel-mamba (d_model=2, L=32, d_inner=4): thread per batch ---------------- */
__global__ void mamba2_kernel(const float* __restrict__ foc,
                              const float* __restrict__ in_w, const float* __restrict__ conv_w,
                              const float* __restrict__ conv_b, const float* __restrict__ x_w,
                              const float* __restrict__ dt_w, const float* __restrict__ dt_b,
                              const float* __restrict__ A_log, const float* __restrict__ Dp,
                              const float* __restrict__ out_w, float* __restrict__ foc_out)
{
    int b = blockIdx.x * blockDim.x + threadIdx.x;
    if (b >= BB) return;
    float xs[CC][DI2], zz[CC][DI2], xcv[CC][DI2];
    for (int l = 0; l < CC; ++l) {
        float f0 = foc[b * CC + l];
        float f1 = foc[b * CC + (CC - 1 - l)];
        for (int j = 0; j < 2 * DI2; ++j) {
            float v = in_w[j * 2 + 0] * f0 + in_w[j * 2 + 1] * f1;
            if (j < DI2) xs[l][j] = v; else zz[l][j - DI2] = v;
        }
    }
    for (int l = 0; l < CC; ++l)
        for (int d = 0; d < DI2; ++d) {
            float acc = conv_b[d];
            for (int k = 0; k < 4; ++k) {
                int li = l - 3 + k;
                if (li >= 0) acc += conv_w[d * 4 + k] * xs[li][d];
            }
            xcv[l][d] = siluf(acc);
        }
    float h[DI2][NST];
    for (int d = 0; d < DI2; ++d) for (int s = 0; s < NST; ++s) h[d][s] = 0.f;
    for (int l = 0; l < CC; ++l) {
        float dtraw = 0.f, Bs[NST], Cs[NST];
        for (int s = 0; s < NST; ++s) { Bs[s] = 0.f; Cs[s] = 0.f; }
        for (int d = 0; d < DI2; ++d) {
            float xv = xcv[l][d];
            dtraw += x_w[d] * xv;   /* row 0 of (33,4) */
            for (int s = 0; s < NST; ++s) {
                Bs[s] += x_w[(1 + s) * DI2 + d] * xv;
                Cs[s] += x_w[(1 + NST + s) * DI2 + d] * xv;
            }
        }
        float yo[DI2];
        for (int d = 0; d < DI2; ++d) {
            float dtv = softplusf(dtraw * dt_w[d] + dt_b[d]);
            float y = 0.f;
            for (int s = 0; s < NST; ++s) {
                float Av = -__expf(A_log[d * NST + s]);
                h[d][s] = __expf(dtv * Av) * h[d][s] + dtv * Bs[s] * xcv[l][d];
                y += h[d][s] * Cs[s];
            }
            yo[d] = (y + Dp[d] * xcv[l][d]) * siluf(zz[l][d]);
        }
        float o0 = 0.f, o1 = 0.f;
        for (int d = 0; d < DI2; ++d) { o0 += out_w[d] * yo[d]; o1 += out_w[DI2 + d] * yo[d]; }
        foc_out[b * CC + l] = o0 + o1;
    }
}

/* ---------------- foss = res*(1+foc2), packed f16 for op GEMM ---------------- */
__global__ void foss_kernel(const float* __restrict__ res, const float* __restrict__ foc2,
                            _Float16* __restrict__ foss16)
{
    int idx = blockIdx.x * blockDim.x + threadIdx.x;
    if (idx >= BB * HW * CC) return;
    int c = idx & 31, b = idx >> 17;
    foss16[idx] = (_Float16)(res[idx] * (1.f + foc2[b * CC + c]));
}

/* ---------------- gated residual (token-major) ---------------- */
__global__ void resid1_kernel(const float* __restrict__ ossout, const float* __restrict__ e1,
                              const float* __restrict__ xt, float* __restrict__ x1)
{
    int idx = blockIdx.x * blockDim.x + threadIdx.x;
    if (idx >= BB * HW * CC) return;
    int c = idx & 31, b = idx >> 17;
    x1[idx] = e1[b * 96 + 64 + c] * ossout[idx] + xt[idx];
}

/* -------- depthwise 3x3 on 128ch, split, exact-GELU(y1)*y2, f16 out (K=64 for pout) -------- */
__global__ void edw_geglu_kernel(const float* __restrict__ yb, const float* __restrict__ ew,
                                 const float* __restrict__ eb, _Float16* __restrict__ g16)
{
    int idx = blockIdx.x * blockDim.x + threadIdx.x;
    if (idx >= BB * HW * HID) return;
    int c = idx & 63, p = (idx >> 6) & (HW - 1), b = idx >> 18;
    int h = p >> 6, w = p & 63;
    float v1 = eb[c], v2 = eb[64 + c];
#pragma unroll
    for (int dy = -1; dy <= 1; ++dy)
#pragma unroll
        for (int dx = -1; dx <= 1; ++dx) {
            int hh = h + dy, ww = w + dx;
            if (hh >= 0 && hh < HH && ww >= 0 && ww < WWD) {
                size_t src = ((size_t)(b * HW) + (hh << 6) + ww) * 128;
                int kidx = (dy + 1) * 3 + (dx + 1);
                v1 += ew[c * 9 + kidx] * yb[src + c];
                v2 += ew[(64 + c) * 9 + kidx] * yb[src + 64 + c];
            }
        }
    float g = 0.5f * v1 * (1.f + erff(v1 * 0.70710678118f));
    g16[idx] = (_Float16)(g * v2);
}

/* ---------------- final: gate2*pout + residual, transpose to (B,C,H,W) ---------------- */
__global__ void final_kernel(const float* __restrict__ pw, const float* __restrict__ e2,
                             const float* __restrict__ x1, float* __restrict__ out)
{
    int idx = blockIdx.x * blockDim.x + threadIdx.x;
    if (idx >= BB * CC * HW) return;
    int p = idx & (HW - 1), c = (idx >> 12) & 31, b = idx >> 17;
    size_t t = ((size_t)(b * HW) + p) * CC + c;
    out[idx] = e2[b * 96 + 64 + c] * pw[t] + x1[t];
}

/* ======================================================================== */
extern "C" void kernel_launch(void* const* d_in, const int* in_sizes, int n_in,
                              void* d_out, int out_size, void* d_ws, size_t ws_size,
                              hipStream_t stream)
{
    (void)in_sizes; (void)n_in; (void)out_size; (void)ws_size;
    const float* x     = (const float*)d_in[0];
    const float* emb   = (const float*)d_in[1];
    const float* a1_w  = (const float*)d_in[2];
    const float* a1_b  = (const float*)d_in[3];
    const float* a2_w  = (const float*)d_in[4];
    const float* a2_b  = (const float*)d_in[5];
    const float* ip_w  = (const float*)d_in[6];
    const float* ip_b  = (const float*)d_in[7];
    const float* dw_w  = (const float*)d_in[8];
    const float* dw_b  = (const float*)d_in[9];
    const float* op_w  = (const float*)d_in[10];
    const float* op_b  = (const float*)d_in[11];
    const float* pin_w = (const float*)d_in[12];
    const float* pin_b = (const float*)d_in[13];
    const float* edw_w = (const float*)d_in[14];
    const float* edw_b = (const float*)d_in[15];
    const float* pout_w= (const float*)d_in[16];
    const float* pout_b= (const float*)d_in[17];
    /* m1 */
    const float* m1_in_w  = (const float*)d_in[18];
    const float* m1_cw    = (const float*)d_in[19];
    const float* m1_cb    = (const float*)d_in[20];
    const float* m1_xw    = (const float*)d_in[21];
    const float* m1_dtw   = (const float*)d_in[22];
    const float* m1_dtb   = (const float*)d_in[23];
    const float* m1_Alog  = (const float*)d_in[24];
    const float* m1_D     = (const float*)d_in[25];
    const float* m1_outw  = (const float*)d_in[26];
    /* m2 */
    const float* m2_in_w  = (const float*)d_in[27];
    const float* m2_cw    = (const float*)d_in[28];
    const float* m2_cb    = (const float*)d_in[29];
    const float* m2_xw    = (const float*)d_in[30];
    const float* m2_dtw   = (const float*)d_in[31];
    const float* m2_dtb   = (const float*)d_in[32];
    const float* m2_Alog  = (const float*)d_in[33];
    const float* m2_D     = (const float*)d_in[34];
    const float* m2_outw  = (const float*)d_in[35];

    float* out = (float*)d_out;

    /* ---- workspace arena ---- */
    char* ws = (char*)d_ws;
    size_t off = 0;
    auto alloc = [&](size_t bytes) -> void* {
        off = (off + 255) & ~(size_t)255;
        void* p = ws + off;
        off += bytes;
        return p;
    };
    float*    e1      = (float*)alloc(BB * 96 * 4);
    float*    e2      = (float*)alloc(BB * 96 * 4);
    float*    xt      = (float*)alloc((size_t)MM * CC * 4);
    _Float16* xn1_16  = (_Float16*)alloc((size_t)MM * CC * 2);
    float*    fo      = (float*)alloc((size_t)MM * 64 * 4);
    _Float16* fop16   = (_Float16*)alloc((size_t)MM * 128 * 2);
    float*    fo2s    = (float*)alloc((size_t)MM * CC * 4);
    float*    xzb     = (float*)alloc((size_t)MM * 512 * 4);
    float*    xc      = (float*)alloc((size_t)MM * DI1 * 4);
    _Float16* xc16    = (_Float16*)alloc((size_t)MM * DI1 * 2);
    float*    projb   = (float*)alloc((size_t)MM * NPROJ1 * 4);
    float*    dtb     = (float*)alloc((size_t)MM * DI1 * 4);
    _Float16* y16     = (_Float16*)alloc((size_t)MM * DI1 * 2);
    float*    mam1    = (float*)alloc((size_t)MM * 128 * 4);
    float*    res     = (float*)alloc((size_t)MM * CC * 4);
    float*    foc     = (float*)alloc(BB * CC * 4);
    float*    foc2    = (float*)alloc(BB * CC * 4);
    _Float16* foss16  = (_Float16*)alloc((size_t)MM * CC * 2);
    float*    ossout  = (float*)alloc((size_t)MM * CC * 4);
    float*    x1      = (float*)alloc((size_t)MM * CC * 4);
    _Float16* xn2_16  = (_Float16*)alloc((size_t)MM * CC * 2);
    float*    yb      = (float*)alloc((size_t)MM * 128 * 4);
    _Float16* g16     = (_Float16*)alloc((size_t)MM * HID * 2);
    float*    pw      = (float*)alloc((size_t)MM * CC * 4);
    /* f16 weights (row counts padded to multiple of 16, zero-filled) */
    _Float16* ip_w16   = (_Float16*)alloc(64 * 32 * 2);
    _Float16* in_w16   = (_Float16*)alloc(512 * 128 * 2);
    _Float16* x_w16    = (_Float16*)alloc(NPROJ1_PAD * 256 * 2);
    _Float16* out_w16  = (_Float16*)alloc(128 * 256 * 2);
    _Float16* op_w16   = (_Float16*)alloc(32 * 32 * 2);
    _Float16* pin_w16  = (_Float16*)alloc(128 * 32 * 2);
    _Float16* pout_w16 = (_Float16*)alloc(32 * 64 * 2);

    auto pack = [&](const float* s, _Float16* d, int n_src, int n_total) {
        pack_f16_kernel<<<(n_total + 255) / 256, 256, 0, stream>>>(s, d, n_src, n_total);
    };
    pack(ip_w,    ip_w16,   64 * 32,       64 * 32);
    pack(m1_in_w, in_w16,   512 * 128,     512 * 128);
    pack(m1_xw,   x_w16,    NPROJ1 * 256,  NPROJ1_PAD * 256);  /* zero-pad rows 40..47 */
    pack(m1_outw, out_w16,  128 * 256,     128 * 256);
    pack(op_w,    op_w16,   32 * 32,       32 * 32);
    pack(pin_w,   pin_w16,  128 * 32,      128 * 32);
    pack(pout_w,  pout_w16, 32 * 64,       32 * 64);

    auto gemm = [&](const _Float16* A, const _Float16* W, const float* bias, float* C,
                    int M, int N, int K) {
        dim3 grid((N + 15) / 16, M / 16);
        wmma_gemm_kernel<<<grid, 32, 0, stream>>>(A, W, bias, C, M, N, K);
    };

    const int EW = 256;  /* generic elementwise block */
    emb_mlp_kernel<<<BB, 192, 0, stream>>>(emb, a1_w, a1_b, a2_w, a2_b, e1, e2);
    transpose_kernel<<<(BB * HW * CC + EW - 1) / EW, EW, 0, stream>>>(x, xt);
    adaln_kernel<<<(MM + 127) / 128, 128, 0, stream>>>(xt, e1, xn1_16);
    gemm(xn1_16, ip_w16, ip_b, fo, MM, 64, 32);                       /* 1x1 in-proj   */
    dw_dir_kernel<<<(BB * HW * CC + EW - 1) / EW, EW, 0, stream>>>(fo, dw_w, dw_b, fop16, fo2s);
    gemm(fop16, in_w16, nullptr, xzb, MM, 512, 128);                  /* mamba in-proj */
    cconv1_kernel<<<(BB * HW * DI1 + EW - 1) / EW, EW, 0, stream>>>(xzb, m1_cw, m1_cb, xc, xc16);
    gemm(xc16, x_w16, nullptr, projb, MM, NPROJ1, 256);               /* x-proj (N=40) */
    dt_kernel<<<(BB * HW * DI1 + EW - 1) / EW, EW, 0, stream>>>(projb, m1_dtw, m1_dtb, dtb);
    mamba1_scan_kernel<<<BB, 256, 0, stream>>>(projb, dtb, xc, xzb, m1_Alog, m1_D, y16);
    gemm(y16, out_w16, nullptr, mam1, MM, 128, 256);                  /* mamba out-proj*/
    dirsum_kernel<<<(BB * HW * CC + EW - 1) / EW, EW, 0, stream>>>(mam1, fo2s, res);
    chmean_kernel<<<BB * CC, 128, 0, stream>>>(res, foc);
    mamba2_kernel<<<1, 32, 0, stream>>>(foc, m2_in_w, m2_cw, m2_cb, m2_xw, m2_dtw,
                                        m2_dtb, m2_Alog, m2_D, m2_outw, foc2);
    foss_kernel<<<(BB * HW * CC + EW - 1) / EW, EW, 0, stream>>>(res, foc2, foss16);
    gemm(foss16, op_w16, op_b, ossout, MM, 32, 32);                   /* 1x1 out-proj  */
    resid1_kernel<<<(BB * HW * CC + EW - 1) / EW, EW, 0, stream>>>(ossout, e1, xt, x1);
    adaln_kernel<<<(MM + 127) / 128, 128, 0, stream>>>(x1, e2, xn2_16);
    gemm(xn2_16, pin_w16, pin_b, yb, MM, 128, 32);                    /* MLP in        */
    edw_geglu_kernel<<<(BB * HW * HID + EW - 1) / EW, EW, 0, stream>>>(yb, edw_w, edw_b, g16);
    gemm(g16, pout_w16, pout_b, pw, MM, 32, 64);                      /* MLP out       */
    final_kernel<<<(BB * CC * HW + EW - 1) / EW, EW, 0, stream>>>(pw, e2, x1, out);
}
